// GAT_35150012351107
// MI455X (gfx1250) — matrix operand order repaired and checked by reading, not compiled, not run on hardware
//
#include <hip/hip_runtime.h>
#include <hip/hip_bf16.h>
#include <limits.h>

#define N_NODES   40000
#define N_EDGES   320000
#define N_GRAPHS  64
#define IN_DIM    300
#define HID       128
#define HEADS     4
#define EDGE_DIM  32
#define FDIM      (HEADS*HID)   /* 512 */
#define SLOPE     0.2f
#define GN_EPS    1e-5f
#define MPAD      40064         /* 313 * 128 */
#define KPAD0     320           /* IN_DIM padded to mult of 32 */
#define KPAD1     128

typedef __attribute__((ext_vector_type(16))) __bf16 v16bf;
typedef __attribute__((ext_vector_type(8)))  float  v8f;

__device__ __forceinline__ unsigned short f2bf(float f) {
    unsigned u = __float_as_uint(f);
    unsigned r = ((u >> 16) & 1u) + 0x7FFFu;   // round-to-nearest-even
    return (unsigned short)((u + r) >> 16);
}
__device__ __forceinline__ int ordKey(float f) {   // order-preserving int encoding
    int i = __float_as_int(f);
    return i >= 0 ? i : (i ^ 0x7FFFFFFF);
}
__device__ __forceinline__ float ordVal(int i) {
    return __int_as_float(i >= 0 ? i : (i ^ 0x7FFFFFFF));
}
__device__ __forceinline__ void wait_async_all() {
#if defined(__has_builtin) && __has_builtin(__builtin_amdgcn_s_wait_asynccnt)
    __builtin_amdgcn_s_wait_asynccnt(0);
#else
    asm volatile("s_wait_asynccnt 0x0" ::: "memory");
#endif
}
// Async global->LDS copy of 16 bytes (ASYNCcnt-tracked).  The LDS address is
// derived from the real LDS pointer (low 32 bits of a generic LDS pointer are
// the LDS byte offset, ISA 10.2) — the ptrtoint feeding the asm captures the
// LDS buffer so the "memory" clobber correctly orders later LDS reads.
__device__ __forceinline__ void async_copy_b128(const unsigned short* lds_elem,
                                                const unsigned short* gaddr) {
    asm volatile("global_load_async_to_lds_b128 %0, %1, off"
                 :: "v"((unsigned)(uintptr_t)lds_elem),
                    "v"((unsigned long long)(uintptr_t)gaddr)
                 : "memory");
}

// ---------------------------------------------------------------------------
// Pre-passes: convert/pad A to bf16 [Mpad][Kpad]; weights to bf16 transposed
// [n=512][Kpad].  Moves all bounds logic + conversion out of the GEMM loop.
// ---------------------------------------------------------------------------
__global__ void cvt_cast_a(const float* __restrict__ A, int M, int K, int Kpad,
                           int total, unsigned short* __restrict__ out)
{
    int i = blockIdx.x * blockDim.x + threadIdx.x;
    if (i >= total) return;
    int m = i / Kpad, k = i - m * Kpad;
    float v = (m < M && k < K) ? A[(size_t)m * K + k] : 0.0f;
    out[i] = f2bf(v);
}

__global__ void cvt_cast_bT(const float* __restrict__ B, int K, int Kpad,
                            unsigned short* __restrict__ outT)
{
    int i = blockIdx.x * blockDim.x + threadIdx.x;
    if (i >= FDIM * Kpad) return;
    int n = i / Kpad, k = i - n * Kpad;
    float v = (k < K) ? B[(size_t)k * FDIM + n] : 0.0f;
    outT[i] = f2bf(v);
}

// ---------------------------------------------------------------------------
// bf16 WMMA GEMM:  C[M x Ntot] = Abf[Mpad x Kpad] * BT^T + bias
// Abf padded bf16 row-major; BT bf16 [Ntot][Kpad] (K-contiguous per column).
// Block: 256 thr (8 waves). Block tile 128(M) x 64(N). Wave: 16(M) x 64(N).
// A tile staged to LDS via async global->LDS copies; B fragments loaded
// straight from the L2-resident weight panel.
// ---------------------------------------------------------------------------
__global__ __launch_bounds__(256)
void gemm_bf16_wmma(const unsigned short* __restrict__ Abf, int M, int Kpad,
                    const unsigned short* __restrict__ BT,
                    const float* __restrict__ bias,
                    float* __restrict__ C, int Ntot)
{
    __shared__ unsigned short sA[128 * 32];   // [row][k]  8 KB

    const int tid    = threadIdx.x;
    const int lane   = tid & 31;
    const int wv     = tid >> 5;
    const int row0   = blockIdx.x * 128;
    const int col0   = blockIdx.y * 64;
    const int halfhi = (lane >> 4) & 1;
    const int mrow   = lane & 15;

    v8f acc[4];
    #pragma unroll
    for (int t = 0; t < 4; ++t) acc[t] = (v8f){0,0,0,0,0,0,0,0};

    const int ksteps = Kpad >> 5;
    for (int ks = 0; ks < ksteps; ++ks) {
        const int k0 = ks << 5;
        // ---- stage A tile: 512 x 16B async transfers, 2 per thread ----
        #pragma unroll
        for (int it = 0; it < 2; ++it) {
            int i  = tid + it * 256;          // b128 chunk id, 0..511
            int r  = i >> 2;                  // 0..127
            int kk = (i & 3) * 8;             // 0,8,16,24
            const unsigned short* g = Abf + (size_t)(row0 + r) * Kpad + k0 + kk;
            async_copy_b128(&sA[r * 32 + kk], g);
        }
        wait_async_all();
        __syncthreads();

        // A fragment (16x32): lanes 0-15 K {0..7,16..23}; lanes 16-31 K {8..15,24..31}
        union { v16bf v; uint4 q[2]; } af;
        {
            int base = (wv * 16 + mrow) * 32;
            af.q[0] = *(const uint4*)&sA[base + (halfhi ? 8 : 0)];
            af.q[1] = *(const uint4*)&sA[base + (halfhi ? 24 : 16)];
        }
        // B fragments straight from global (weight panel stays in L2)
        #pragma unroll
        for (int t = 0; t < 4; ++t) {
            union { v16bf v; uint4 q[2]; } bfm;
            const unsigned short* bp =
                BT + (size_t)(col0 + t * 16 + mrow) * Kpad + k0 + (halfhi ? 8 : 0);
            bfm.q[0] = *(const uint4*)bp;
            bfm.q[1] = *(const uint4*)(bp + 16);
            acc[t] = __builtin_amdgcn_wmma_f32_16x16x32_bf16(
                         false, af.v, false, bfm.v, (short)0, acc[t], false, false);
        }
        __syncthreads();
    }

    // Epilogue: C/D layout — VGPR i: lanes 0-15 => M=i, lanes 16-31 => M=i+8; N = lane&15
    #pragma unroll
    for (int t = 0; t < 4; ++t) {
        int col = col0 + t * 16 + mrow;
        float bv = bias ? bias[col] : 0.0f;
        #pragma unroll
        for (int i = 0; i < 8; ++i) {
            int row = row0 + wv * 16 + i + (halfhi ? 8 : 0);
            if (row < M) C[(size_t)row * Ntot + col] = acc[t][i] + bv;
        }
    }
}

// ---------------------------------------------------------------------------
// Per-layer scratch init
// ---------------------------------------------------------------------------
__global__ void init_layer(int* __restrict__ amaxEnc, float* __restrict__ den,
                           float* __restrict__ hacc, float* __restrict__ gsum,
                           float* __restrict__ gvar, float* __restrict__ gcnt)
{
    int i = blockIdx.x * blockDim.x + threadIdx.x;
    if (i < N_NODES * HID)    hacc[i] = 0.0f;
    if (i < N_NODES * HEADS) { den[i] = 0.0f; amaxEnc[i] = INT_MIN; }
    if (i < N_GRAPHS * HID)  { gsum[i] = 0.0f; gvar[i] = 0.0f; }
    if (i < N_GRAPHS)          gcnt[i] = 0.0f;
}

// ---------------------------------------------------------------------------
// Edge attention logits: one edge per wave.
// em = ea[e] @ We computed on the fly (We cached as bf16 in LDS, 32 KB).
// ---------------------------------------------------------------------------
__global__ __launch_bounds__(256)
void edge_alpha(const float* __restrict__ xl, const float* __restrict__ xr,
                const float* __restrict__ ea, const int* __restrict__ src,
                const int* __restrict__ dst, const float* __restrict__ We,
                const float* __restrict__ att, float* __restrict__ alpha,
                int* __restrict__ amaxEnc)
{
    __shared__ unsigned short sWe[EDGE_DIM * FDIM];   // 32 KB
    __shared__ float sAtt[FDIM];

    const int tid = threadIdx.x, lane = tid & 31, wv = tid >> 5;
    for (int i = tid; i < EDGE_DIM * FDIM; i += 256) sWe[i] = f2bf(We[i]);
    for (int i = tid; i < FDIM; i += 256)            sAtt[i] = att[i];
    __syncthreads();

    for (int base = blockIdx.x * 8; base < N_EDGES; base += gridDim.x * 8) {
        const int e = base + wv;
        if (e >= N_EDGES) continue;                 // wave-uniform predicate
        const int s = src[e], d = dst[e];
        const float eav = ea[(size_t)e * EDGE_DIM + lane];

        float em[16];
        #pragma unroll
        for (int j = 0; j < 16; ++j) em[j] = 0.0f;
        #pragma unroll
        for (int k = 0; k < EDGE_DIM; ++k) {
            const float a = __shfl(eav, k, 32);
            const uint4* p = (const uint4*)&sWe[k * FDIM + lane * 16];
            uint4 q0 = p[0], q1 = p[1];
            unsigned w[8] = {q0.x, q0.y, q0.z, q0.w, q1.x, q1.y, q1.z, q1.w};
            #pragma unroll
            for (int j = 0; j < 8; ++j) {
                em[2*j]   += a * __uint_as_float(w[j] << 16);
                em[2*j+1] += a * __uint_as_float(w[j] & 0xFFFF0000u);
            }
        }

        const float4* pxl = (const float4*)&xl[(size_t)s * FDIM + lane * 16];
        const float4* pxr = (const float4*)&xr[(size_t)d * FDIM + lane * 16];
        const float4* pat = (const float4*)&sAtt[lane * 16];
        float partial = 0.0f;
        #pragma unroll
        for (int v = 0; v < 4; ++v) {
            float4 xa = pxl[v], xb = pxr[v], at = pat[v];
            float m;
            m = xa.x + xb.x + em[4*v+0]; m = m >= 0.f ? m : SLOPE * m; partial += m * at.x;
            m = xa.y + xb.y + em[4*v+1]; m = m >= 0.f ? m : SLOPE * m; partial += m * at.y;
            m = xa.z + xb.z + em[4*v+2]; m = m >= 0.f ? m : SLOPE * m; partial += m * at.z;
            m = xa.w + xb.w + em[4*v+3]; m = m >= 0.f ? m : SLOPE * m; partial += m * at.w;
        }
        partial += __shfl_down(partial, 4, 8);
        partial += __shfl_down(partial, 2, 8);
        partial += __shfl_down(partial, 1, 8);
        if ((lane & 7) == 0) {
            const int h = lane >> 3;
            alpha[(size_t)e * HEADS + h] = partial;
            atomicMax(&amaxEnc[d * HEADS + h], ordKey(partial));
        }
    }
}

// ---------------------------------------------------------------------------
// ex = exp(alpha - amax[dst]) (in place); den[dst] += ex
// ---------------------------------------------------------------------------
__global__ void edge_exp(float* __restrict__ alpha, const int* __restrict__ dst,
                         const int* __restrict__ amaxEnc, float* __restrict__ den)
{
    int i = blockIdx.x * blockDim.x + threadIdx.x;
    if (i >= N_EDGES * HEADS) return;
    int e = i >> 2, h = i & 3;
    int d = dst[e];
    float mf = ordVal(amaxEnc[d * HEADS + h]);
    if ((__float_as_uint(mf) & 0x7F800000u) == 0x7F800000u) mf = 0.0f; // non-finite -> 0
    float ex = expf(alpha[i] - mf);
    alpha[i] = ex;
    atomicAdd(&den[d * HEADS + h], ex);
}

// ---------------------------------------------------------------------------
// hacc[dst,c] += sum_h (ex/(den+1e-16)) * xl[src,h,c] ; one edge per wave
// ---------------------------------------------------------------------------
__global__ __launch_bounds__(256)
void edge_scatter(const float* __restrict__ ex, const float* __restrict__ den,
                  const float* __restrict__ xl, const int* __restrict__ src,
                  const int* __restrict__ dst, float* __restrict__ hacc)
{
    const int tid = threadIdx.x, lane = tid & 31, wv = tid >> 5;
    for (int base = blockIdx.x * 8; base < N_EDGES; base += gridDim.x * 8) {
        const int e = base + wv;
        if (e >= N_EDGES) continue;
        const int s = src[e], d = dst[e];
        float a[HEADS];
        #pragma unroll
        for (int h = 0; h < HEADS; ++h)
            a[h] = ex[(size_t)e * HEADS + h] / (den[d * HEADS + h] + 1e-16f);
        const int c = lane * 4;
        float4 accv = {0, 0, 0, 0};
        #pragma unroll
        for (int h = 0; h < HEADS; ++h) {
            float4 xv = *(const float4*)&xl[(size_t)s * FDIM + h * HID + c];
            accv.x += a[h] * xv.x; accv.y += a[h] * xv.y;
            accv.z += a[h] * xv.z; accv.w += a[h] * xv.w;
        }
        atomicAdd(&hacc[(size_t)d * HID + c + 0], accv.x);
        atomicAdd(&hacc[(size_t)d * HID + c + 1], accv.y);
        atomicAdd(&hacc[(size_t)d * HID + c + 2], accv.z);
        atomicAdd(&hacc[(size_t)d * HID + c + 3], accv.w);
    }
}

// ---------------------------------------------------------------------------
// GraphNorm passes
// ---------------------------------------------------------------------------
__global__ void gn_pass1(const float* __restrict__ hacc, const float* __restrict__ bias,
                         const int* __restrict__ batch, float* __restrict__ gsum,
                         float* __restrict__ gcnt, float* __restrict__ hbuf)
{
    int i = blockIdx.x * blockDim.x + threadIdx.x;
    if (i >= N_NODES * HID) return;
    int n = i >> 7, c = i & 127;
    float hv = hacc[i] * (1.0f / HEADS) + bias[c];   // head-mean + bias
    hbuf[i] = hv;
    int g = batch[n];
    atomicAdd(&gsum[g * HID + c], hv);
    if (c == 0) atomicAdd(&gcnt[g], 1.0f);
}

__global__ void gn_pass2(float* __restrict__ hbuf, const int* __restrict__ batch,
                         const float* __restrict__ gsum, const float* __restrict__ gcnt,
                         const float* __restrict__ gms, float* __restrict__ gvar)
{
    int i = blockIdx.x * blockDim.x + threadIdx.x;
    if (i >= N_NODES * HID) return;
    int n = i >> 7, c = i & 127, g = batch[n];
    float mean = gsum[g * HID + c] / gcnt[g];
    float out  = hbuf[i] - mean * gms[c];
    hbuf[i] = out;
    atomicAdd(&gvar[g * HID + c], out * out);
}

__global__ void gn_pass3(float* __restrict__ hbuf, const int* __restrict__ batch,
                         const float* __restrict__ gvar, const float* __restrict__ gcnt,
                         const float* __restrict__ gw, const float* __restrict__ gb,
                         const float* __restrict__ pa)
{
    int i = blockIdx.x * blockDim.x + threadIdx.x;
    if (i >= N_NODES * HID) return;
    int n = i >> 7, c = i & 127, g = batch[n];
    float var = gvar[g * HID + c] / gcnt[g];
    float h = gw[c] * hbuf[i] / sqrtf(var + GN_EPS) + gb[c];
    h = h >= 0.f ? h : pa[c] * h;                     // PReLU
    hbuf[i] = h;
}

// ---------------------------------------------------------------------------
// Segment-max pooling + MLP head
// ---------------------------------------------------------------------------
__global__ void pool_init(int* __restrict__ pooledEnc)
{
    int i = blockIdx.x * blockDim.x + threadIdx.x;
    if (i < N_GRAPHS * HID) pooledEnc[i] = INT_MIN;
}

__global__ void pool_max(const float* __restrict__ hbuf, const int* __restrict__ batch,
                         int* __restrict__ pooledEnc)
{
    int i = blockIdx.x * blockDim.x + threadIdx.x;
    if (i >= N_NODES * HID) return;
    int n = i >> 7, c = i & 127, g = batch[n];
    atomicMax(&pooledEnc[g * HID + c], ordKey(hbuf[i]));
}

__global__ __launch_bounds__(128)
void head_mlp(const int* __restrict__ pooledEnc, const float* __restrict__ Wh1,
              const float* __restrict__ bh1, const float* __restrict__ prh,
              const float* __restrict__ Wh2, const float* __restrict__ bh2,
              float* __restrict__ out)
{
    __shared__ float sp[HID];
    __shared__ float sz[HID];
    const int g = blockIdx.x, c = threadIdx.x;
    sp[c] = ordVal(pooledEnc[g * HID + c]);
    __syncthreads();
    float acc = bh1[c];
    for (int k = 0; k < HID; ++k) acc += sp[k] * Wh1[k * HID + c];
    float z = acc >= 0.f ? acc : prh[c] * acc;
    sz[c] = z * Wh2[c];
    __syncthreads();
    for (int s2 = 64; s2 > 0; s2 >>= 1) {
        if (c < s2) sz[c] += sz[c + s2];
        __syncthreads();
    }
    if (c == 0) out[g] = 1.0f / (1.0f + expf(-(sz[0] + bh2[0])));
}

// ---------------------------------------------------------------------------
// Host orchestration
// ---------------------------------------------------------------------------
extern "C" void kernel_launch(void* const* d_in, const int* in_sizes, int n_in,
                              void* d_out, int out_size, void* d_ws, size_t ws_size,
                              hipStream_t stream)
{
    (void)in_sizes; (void)n_in; (void)out_size; (void)ws_size;

    const float* x     = (const float*)d_in[0];
    const int*   ei    = (const int*)d_in[1];
    const float* ea    = (const float*)d_in[2];
    const int*   batch = (const int*)d_in[3];
    const int* src = ei;
    const int* dst = ei + N_EDGES;

    float* ws = (float*)d_ws;
    size_t off = 0;
    float* xl     = ws + off; off += (size_t)N_NODES * FDIM;
    float* xr     = ws + off; off += (size_t)N_NODES * FDIM;
    float* alpha  = ws + off; off += (size_t)N_EDGES * HEADS;
    int*   amax   = (int*)(ws + off); off += (size_t)N_NODES * HEADS;
    float* den    = ws + off; off += (size_t)N_NODES * HEADS;
    float* hacc   = ws + off; off += (size_t)N_NODES * HID;
    float* hbuf   = ws + off; off += (size_t)N_NODES * HID;
    float* gsum   = ws + off; off += (size_t)N_GRAPHS * HID;
    float* gvar   = ws + off; off += (size_t)N_GRAPHS * HID;
    float* gcnt   = ws + off; off += (size_t)N_GRAPHS;
    int*   pooled = (int*)(ws + off); off += (size_t)N_GRAPHS * HID;
    unsigned short* xbf = (unsigned short*)(ws + off); off += ((size_t)MPAD * KPAD0) / 2;
    unsigned short* wbf = (unsigned short*)(ws + off); off += ((size_t)FDIM * KPAD0) / 2;

    const dim3 gemmGrid((N_NODES + 127) / 128, FDIM / 64);
    const int nhBlocks = (N_NODES * HID + 255) / 256;
    const int ehBlocks = (N_EDGES * HEADS + 255) / 256;

    for (int layer = 0; layer < 2; ++layer) {
        const int p = 4 + layer * 11;
        const float* Wl  = (const float*)d_in[p + 0];
        const float* bl  = (const float*)d_in[p + 1];
        const float* Wr  = (const float*)d_in[p + 2];
        const float* br  = (const float*)d_in[p + 3];
        const float* We  = (const float*)d_in[p + 4];
        const float* att = (const float*)d_in[p + 5];
        const float* bias= (const float*)d_in[p + 6];
        const float* gnw = (const float*)d_in[p + 7];
        const float* gnb = (const float*)d_in[p + 8];
        const float* gms = (const float*)d_in[p + 9];
        const float* pr  = (const float*)d_in[p + 10];

        const float* hin = layer ? hbuf : x;
        const int K    = layer ? HID : IN_DIM;
        const int Kpad = layer ? KPAD1 : KPAD0;
        const int aTot = MPAD * Kpad;

        cvt_cast_a<<<(aTot + 255) / 256, 256, 0, stream>>>(hin, N_NODES, K, Kpad, aTot, xbf);
        cvt_cast_bT<<<(FDIM * Kpad + 255) / 256, 256, 0, stream>>>(Wl, K, Kpad, wbf);
        gemm_bf16_wmma<<<gemmGrid, 256, 0, stream>>>(xbf, N_NODES, Kpad, wbf, bl, xl, FDIM);
        cvt_cast_bT<<<(FDIM * Kpad + 255) / 256, 256, 0, stream>>>(Wr, K, Kpad, wbf);
        gemm_bf16_wmma<<<gemmGrid, 256, 0, stream>>>(xbf, N_NODES, Kpad, wbf, br, xr, FDIM);

        init_layer<<<nhBlocks, 256, 0, stream>>>(amax, den, hacc, gsum, gvar, gcnt);
        edge_alpha<<<4096, 256, 0, stream>>>(xl, xr, ea, src, dst, We, att, alpha, amax);
        edge_exp<<<ehBlocks, 256, 0, stream>>>(alpha, dst, amax, den);
        edge_scatter<<<4096, 256, 0, stream>>>(alpha, den, xl, src, dst, hacc);
        gn_pass1<<<nhBlocks, 256, 0, stream>>>(hacc, bias, batch, gsum, gcnt, hbuf);
        gn_pass2<<<nhBlocks, 256, 0, stream>>>(hbuf, batch, gsum, gcnt, gms, gvar);
        gn_pass3<<<nhBlocks, 256, 0, stream>>>(hbuf, batch, gvar, gcnt, gnw, gnb, pr);
    }

    pool_init<<<(N_GRAPHS * HID + 255) / 256, 256, 0, stream>>>(pooled);
    pool_max<<<nhBlocks, 256, 0, stream>>>(hbuf, batch, pooled);
    head_mlp<<<N_GRAPHS, 128, 0, stream>>>(pooled,
        (const float*)d_in[26], (const float*)d_in[27], (const float*)d_in[28],
        (const float*)d_in[29], (const float*)d_in[30], (float*)d_out);
}
